// DeepARModel_10471130267970
// MI455X (gfx1250) — matrix operand order repaired and checked by reading, not compiled, not run on hardware
//
#include <hip/hip_runtime.h>
#include <math.h>

typedef __attribute__((ext_vector_type(16))) __bf16 v16bf;
typedef __attribute__((ext_vector_type(8)))  __bf16 bf8;
typedef __attribute__((ext_vector_type(8)))  float  v8f;

#define B_   512
#define T_   512
#define P_   64
#define E_   64
#define H_   256
#define NTF_ 8
#define NSF_ 16
#define G4H  1024   // 4*H

// ---------------------------------------------------------------------------
// Weight fusion: U[n] = W_lag @ Wih0[0:64, n];  V[k,n] = W_time[k,:] @ Wih0[64:128, n]
// Sw[j,n] = W_stat[j,:] @ Wih0[128:192, n];  biasAll[n] = b0 + folded embedding biases
// ---------------------------------------------------------------------------
__global__ void fuse_weights(const float* __restrict__ Wlag, const float* __restrict__ blag,
                             const float* __restrict__ Wtime, const float* __restrict__ btime,
                             const float* __restrict__ Wstat, const float* __restrict__ bstat,
                             const float* __restrict__ Wih0, const float* __restrict__ b0,
                             float* __restrict__ U, float* __restrict__ V,
                             float* __restrict__ Sw, float* __restrict__ biasAll) {
    int n = blockIdx.x * blockDim.x + threadIdx.x;
    if (n >= G4H) return;
    float u = 0.f, bacc = b0[n];
    for (int e = 0; e < E_; ++e) {
        float w = Wih0[e * G4H + n];
        u += Wlag[e] * w;
        bacc += blag[e] * w;
    }
    U[n] = u;
    for (int k = 0; k < NTF_; ++k) {
        float acc = 0.f;
        for (int e = 0; e < E_; ++e) acc += Wtime[k * E_ + e] * Wih0[(E_ + e) * G4H + n];
        V[k * G4H + n] = acc;
    }
    for (int e = 0; e < E_; ++e) bacc += btime[e] * Wih0[(E_ + e) * G4H + n];
    for (int j = 0; j < NSF_; ++j) {
        float acc = 0.f;
        for (int e = 0; e < E_; ++e) acc += Wstat[j * E_ + e] * Wih0[(2 * E_ + e) * G4H + n];
        Sw[j * G4H + n] = acc;
    }
    for (int e = 0; e < E_; ++e) bacc += bstat[e] * Wih0[(2 * E_ + e) * G4H + n];
    biasAll[n] = bacc;
}

// Scon[b,n] = static[b,:] @ Sw[:,n] + biasAll[n]
__global__ void compute_scon(const float* __restrict__ stat, const float* __restrict__ Sw,
                             const float* __restrict__ biasAll, float* __restrict__ Scon) {
    int n = blockIdx.y * blockDim.x + threadIdx.x;
    int b = blockIdx.x;
    float acc = biasAll[n];
    for (int j = 0; j < NSF_; ++j) acc += stat[b * NSF_ + j] * Sw[j * G4H + n];
    Scon[b * G4H + n] = acc;
}

// (K=256, N=1024) fp32 row-major -> (N=1024, K=256) bf16  (K-pairs contiguous for B-operand)
__global__ void transpose_to_bf16(const float* __restrict__ W, __bf16* __restrict__ WT) {
    int i = blockIdx.x * blockDim.x + threadIdx.x;   // 0 .. 256*1024-1
    int k = i >> 10;
    int n = i & 1023;
    WT[n * H_ + k] = (__bf16)W[i];
}

__global__ void init_state(__bf16* h0a, __bf16* h0b, __bf16* h1a, __bf16* h1b,
                           float* c0, float* c1, float* h1f, float* cur_y,
                           const float* __restrict__ y) {
    int i = blockIdx.x * blockDim.x + threadIdx.x;   // exactly B_*H_ threads
    __bf16 zb = (__bf16)0.f;
    h0a[i] = zb; h0b[i] = zb; h1a[i] = zb; h1b[i] = zb;
    c0[i] = 0.f; c1[i] = 0.f; h1f[i] = 0.f;
    if (i < B_) cur_y[i] = y[i * T_ + (T_ - 1)];
}

__device__ __forceinline__ float sigm(float x) { return 1.f / (1.f + __expf(-x)); }

// Cooperative stage of one gate-panel: 4 gates x 16 n-rows x 256 k of bf16 (32 KB).
// LDS layout: row index (g*16 + j), 256 bf16 per row -> lane reads are 32B contiguous.
__device__ __forceinline__ void stage_panel(uint4* lb4, const __bf16* WT,
                                            int hTile, int tid) {
    const uint4* ws4 = (const uint4*)WT;
    for (int c = tid; c < 2048; c += 256) {
        int row = c >> 5, off = c & 31;                       // 32 x 16B per row
        int n = (row >> 4) * H_ + hTile * 16 + (row & 15);    // global WT row
        lb4[c] = ws4[n * 32 + off];
    }
}

// ---------------------------------------------------------------------------
// Layer-0 step: z = h0_prev @ Whh0 (WMMA K=256, B-panel via LDS) + [y*U + tf@V + Scon]
//               -> gates -> h0, c0.   WG: 8 waves, all on one hTile (128 rows x 16 h).
// ---------------------------------------------------------------------------
__global__ __launch_bounds__(256) void lstm_l0_step(
    const __bf16* __restrict__ hsrc, __bf16* __restrict__ hdst,
    float* __restrict__ c0,
    const __bf16* __restrict__ WhhT,
    const float* __restrict__ U, const float* __restrict__ V,
    const float* __restrict__ Scon,
    const float* __restrict__ ybuf,     // y (cond) or cur_y (pred)
    const float* __restrict__ tfb,      // time_features or future_time_features
    int t, int mode) {                  // mode 0 = conditioning, 1 = prediction
    __shared__ __align__(32) uint4 lb4[2048];     // 32 KB weight panel
    const int tid   = threadIdx.x;
    const int lane  = tid & 31;
    const int wave  = tid >> 5;
    const int hTile = blockIdx.y;                 // 0..15
    const int mTile = blockIdx.x * 8 + wave;      // 0..31
    const int nlo   = lane & 15;
    const int up    = lane >> 4;

    stage_panel(lb4, WhhT, hTile, tid);
    __syncthreads();
    const __bf16* lb = (const __bf16*)lb4;

    const v8f vz = {0.f, 0.f, 0.f, 0.f, 0.f, 0.f, 0.f, 0.f};
    v8f acc[4] = {vz, vz, vz, vz};
    const __bf16* arow = hsrc + (mTile * 16 + nlo) * H_;

    for (int kb = 0; kb < H_; kb += 32) {
        union { v16bf v; bf8 h[2]; } A;                        // contiguous k = 16*up + s (matches B)
        A.h[0] = *(const bf8*)(arow + kb + up * 16);
        A.h[1] = *(const bf8*)(arow + kb + up * 16 + 8);
#pragma unroll
        for (int g = 0; g < 4; ++g) {                          // contiguous k = 16*up + s (matches A)
            v16bf Bv = *(const v16bf*)(lb + (g * 16 + nlo) * H_ + kb + up * 16);
            acc[g] = __builtin_amdgcn_wmma_f32_16x16x32_bf16(
                false, A.v, false, Bv, (short)0, acc[g], false, false);
        }
    }

    const int hcol = hTile * 16 + nlo;
    float Ug[4];
    float Vg[4][NTF_];
#pragma unroll
    for (int g = 0; g < 4; ++g) {
        int n = g * H_ + hcol;
        Ug[g] = U[n];
#pragma unroll
        for (int k = 0; k < NTF_; ++k) Vg[g][k] = V[k * G4H + n];
    }
    const int mbase = mTile * 16 + up * 8;
#pragma unroll
    for (int v = 0; v < 8; ++v) {
        int b = mbase + v;
        float yv = mode ? ybuf[b] : (t == 0 ? 0.f : ybuf[b * T_ + t - 1]);
        const float* tf = mode ? (tfb + (b * P_ + t) * NTF_)
                               : (tfb + (b * T_ + t) * NTF_);
        float tfl[NTF_];
#pragma unroll
        for (int k = 0; k < NTF_; ++k) tfl[k] = tf[k];
        float z[4];
#pragma unroll
        for (int g = 0; g < 4; ++g) {
            float x = acc[g][v] + Ug[g] * yv + Scon[b * G4H + g * H_ + hcol];
#pragma unroll
            for (int k = 0; k < NTF_; ++k) x += tfl[k] * Vg[g][k];
            z[g] = x;
        }
        float cprev = c0[b * H_ + hcol];
        float cn = sigm(z[1]) * cprev + sigm(z[0]) * tanhf(z[2]);
        float hn = sigm(z[3]) * tanhf(cn);
        c0[b * H_ + hcol]   = cn;
        hdst[b * H_ + hcol] = (__bf16)hn;
    }
}

// ---------------------------------------------------------------------------
// Layer-1 step: z = h0 @ Wih1 + h1_prev @ Whh1 + b1 -> gates -> h1, c1 (+f32 h1).
// Two LDS passes over the same 32 KB buffer: Wih1 panel (A=h0), then Whh1 (A=h1).
// ---------------------------------------------------------------------------
__global__ __launch_bounds__(256) void lstm_l1_step(
    const __bf16* __restrict__ h0cur, const __bf16* __restrict__ h1src,
    __bf16* __restrict__ h1dst, float* __restrict__ h1f, float* __restrict__ c1,
    const __bf16* __restrict__ WihT, const __bf16* __restrict__ WhhT,
    const float* __restrict__ bias) {
    __shared__ __align__(32) uint4 lb4[2048];     // 32 KB weight panel (reused per pass)
    const int tid   = threadIdx.x;
    const int lane  = tid & 31;
    const int wave  = tid >> 5;
    const int hTile = blockIdx.y;
    const int mTile = blockIdx.x * 8 + wave;
    const int nlo   = lane & 15;
    const int up    = lane >> 4;

    const v8f vz = {0.f, 0.f, 0.f, 0.f, 0.f, 0.f, 0.f, 0.f};
    v8f acc[4] = {vz, vz, vz, vz};
    const __bf16* lb = (const __bf16*)lb4;

#pragma unroll 1
    for (int pass = 0; pass < 2; ++pass) {
        if (pass) __syncthreads();                 // all waves done with previous panel
        stage_panel(lb4, pass ? WhhT : WihT, hTile, tid);
        __syncthreads();
        const __bf16* arow = (pass ? h1src : h0cur) + (mTile * 16 + nlo) * H_;
        for (int kb = 0; kb < H_; kb += 32) {
            union { v16bf v; bf8 h[2]; } A;
            A.h[0] = *(const bf8*)(arow + kb + up * 16);
            A.h[1] = *(const bf8*)(arow + kb + up * 16 + 8);
#pragma unroll
            for (int g = 0; g < 4; ++g) {
                v16bf Bv = *(const v16bf*)(lb + (g * 16 + nlo) * H_ + kb + up * 16);
                acc[g] = __builtin_amdgcn_wmma_f32_16x16x32_bf16(
                    false, A.v, false, Bv, (short)0, acc[g], false, false);
            }
        }
    }

    const int hcol = hTile * 16 + nlo;
    float bg[4];
#pragma unroll
    for (int g = 0; g < 4; ++g) bg[g] = bias[g * H_ + hcol];
    const int mbase = mTile * 16 + up * 8;
#pragma unroll
    for (int v = 0; v < 8; ++v) {
        int b = mbase + v;
        float z[4];
#pragma unroll
        for (int g = 0; g < 4; ++g) z[g] = acc[g][v] + bg[g];
        float cprev = c1[b * H_ + hcol];
        float cn = sigm(z[1]) * cprev + sigm(z[0]) * tanhf(z[2]);
        float hn = sigm(z[3]) * tanhf(cn);
        c1[b * H_ + hcol]    = cn;
        h1f[b * H_ + hcol]   = hn;
        h1dst[b * H_ + hcol] = (__bf16)hn;
    }
}

// ---------------------------------------------------------------------------
// Head: mu/sigma/sample for one prediction step; feeds sample back as cur_y.
// ---------------------------------------------------------------------------
__global__ void head_step(const float* __restrict__ h1f,
                          const float* __restrict__ Wmu, const float* __restrict__ bmu,
                          const float* __restrict__ Wsig, const float* __restrict__ bsig,
                          const float* __restrict__ eps,
                          float* __restrict__ out, float* __restrict__ cur_y, int t) {
    int b = blockIdx.x * blockDim.x + threadIdx.x;
    if (b >= B_) return;
    const float* h = h1f + b * H_;
    float mu = bmu[0], sg = bsig[0];
    for (int k = 0; k < H_; ++k) {
        float hv = h[k];
        mu += hv * Wmu[k];
        sg += hv * Wsig[k];
    }
    float sp = (sg > 20.f) ? sg : log1pf(__expf(sg));
    float sigma = sp + 1e-5f;
    float samp = mu + sigma * eps[b * P_ + t];
    out[0 * B_ * P_ + b * P_ + t] = mu;
    out[1 * B_ * P_ + b * P_ + t] = sigma;
    out[2 * B_ * P_ + b * P_ + t] = samp;
    cur_y[b] = samp;
}

// ---------------------------------------------------------------------------
extern "C" void kernel_launch(void* const* d_in, const int* in_sizes, int n_in,
                              void* d_out, int out_size, void* d_ws, size_t ws_size,
                              hipStream_t stream) {
    (void)in_sizes; (void)n_in; (void)out_size; (void)ws_size;
    const float* y     = (const float*)d_in[0];
    const float* tf    = (const float*)d_in[1];
    const float* stat  = (const float*)d_in[2];
    const float* ftf   = (const float*)d_in[3];
    const float* eps   = (const float*)d_in[4];
    const float* Wlag  = (const float*)d_in[5];
    const float* blag  = (const float*)d_in[6];
    const float* Wtime = (const float*)d_in[7];
    const float* btime = (const float*)d_in[8];
    const float* Wstat = (const float*)d_in[9];
    const float* bstat = (const float*)d_in[10];
    const float* Wih0  = (const float*)d_in[11];
    const float* Whh0  = (const float*)d_in[12];
    const float* b0    = (const float*)d_in[13];
    const float* Wih1  = (const float*)d_in[14];
    const float* Whh1  = (const float*)d_in[15];
    const float* b1    = (const float*)d_in[16];
    const float* Wmu   = (const float*)d_in[17];
    const float* bmu   = (const float*)d_in[18];
    const float* Wsig  = (const float*)d_in[19];
    const float* bsig  = (const float*)d_in[20];
    float* out = (float*)d_out;

    // workspace carve-up (~6.3 MB total)
    char* p = (char*)d_ws;
    auto carve = [&](size_t bytes) -> char* {
        char* r = p;
        p += (bytes + 255) & ~(size_t)255;
        return r;
    };
    __bf16* h0buf[2] = {(__bf16*)carve(B_ * H_ * 2), (__bf16*)carve(B_ * H_ * 2)};
    __bf16* h1buf[2] = {(__bf16*)carve(B_ * H_ * 2), (__bf16*)carve(B_ * H_ * 2)};
    float*  h1f     = (float*)carve(B_ * H_ * 4);
    float*  c0p     = (float*)carve(B_ * H_ * 4);
    float*  c1p     = (float*)carve(B_ * H_ * 4);
    float*  Scon    = (float*)carve(B_ * G4H * 4);
    float*  U       = (float*)carve(G4H * 4);
    float*  V       = (float*)carve(NTF_ * G4H * 4);
    float*  Sw      = (float*)carve(NSF_ * G4H * 4);
    float*  biasAll = (float*)carve(G4H * 4);
    __bf16* WhhT0   = (__bf16*)carve(G4H * H_ * 2);
    __bf16* WihT1   = (__bf16*)carve(G4H * H_ * 2);
    __bf16* WhhT1   = (__bf16*)carve(G4H * H_ * 2);
    float*  cur_y   = (float*)carve(B_ * 4);

    // preprocessing
    fuse_weights<<<4, 256, 0, stream>>>(Wlag, blag, Wtime, btime, Wstat, bstat,
                                        Wih0, b0, U, V, Sw, biasAll);
    compute_scon<<<dim3(B_, 4), 256, 0, stream>>>(stat, Sw, biasAll, Scon);
    transpose_to_bf16<<<1024, 256, 0, stream>>>(Whh0, WhhT0);
    transpose_to_bf16<<<1024, 256, 0, stream>>>(Wih1, WihT1);
    transpose_to_bf16<<<1024, 256, 0, stream>>>(Whh1, WhhT1);
    init_state<<<B_ * H_ / 256, 256, 0, stream>>>(h0buf[0], h0buf[1], h1buf[0], h1buf[1],
                                                  c0p, c1p, h1f, cur_y, y);

    const dim3 gStep(4, 16);   // 64 WGs x 8 waves: (512 batch) x (256 h) x 4 gates
    int s = 0;

    // conditioning scan
    for (int t = 0; t < T_; ++t) {
        lstm_l0_step<<<gStep, 256, 0, stream>>>(h0buf[s], h0buf[1 - s], c0p, WhhT0,
                                                U, V, Scon, y, tf, t, 0);
        lstm_l1_step<<<gStep, 256, 0, stream>>>(h0buf[1 - s], h1buf[s], h1buf[1 - s],
                                                h1f, c1p, WihT1, WhhT1, b1);
        s ^= 1;
    }
    // prediction scan
    for (int t = 0; t < P_; ++t) {
        lstm_l0_step<<<gStep, 256, 0, stream>>>(h0buf[s], h0buf[1 - s], c0p, WhhT0,
                                                U, V, Scon, cur_y, ftf, t, 1);
        lstm_l1_step<<<gStep, 256, 0, stream>>>(h0buf[1 - s], h1buf[s], h1buf[1 - s],
                                                h1f, c1p, WihT1, WhhT1, b1);
        head_step<<<2, 256, 0, stream>>>(h1f, Wmu, bmu, Wsig, bsig, eps, out, cur_y, t);
        s ^= 1;
    }
}
